// VQVAE_72550587564067
// MI455X (gfx1250) — matrix operand order repaired and checked by reading, not compiled, not run on hardware
//
#include <hip/hip_runtime.h>
#include <hip/hip_bf16.h>
#include <math.h>

typedef unsigned short u16;
typedef __attribute__((ext_vector_type(16))) __bf16 v16bf;
typedef __attribute__((ext_vector_type(8)))  float  v8f;
typedef __attribute__((ext_vector_type(2)))  float  v2f;

union FragBF { unsigned int u[8]; v16bf v; };

__device__ __forceinline__ u16 f2bf(float f){
  unsigned int u = __float_as_uint(f);
  u += 0x7FFFu + ((u >> 16) & 1u);        // round-to-nearest-even
  return (u16)(u >> 16);
}

// ---------------------------------------------------------------- utilities
__global__ void zero_kernel(float* sumsq, int* counts){
  int t = threadIdx.x;
  if (t == 0) sumsq[0] = 0.f;
  if (t < 512) counts[t] = 0;
}

__global__ void prep_bf16_kernel(const float* __restrict__ in, u16* __restrict__ out,
                                 int n, int relu){
  int t = blockIdx.x * 256 + threadIdx.x;
  if (t >= n) return;
  float v = in[t];
  if (relu) v = fmaxf(v, 0.f);
  out[t] = f2bf(v);
}

// ------------------------------------------------- spectral norm: 1/sigma
__global__ void sn_sigma_kernel(const float* __restrict__ w, int rows, int cols,
                                float* __restrict__ inv_sigma){
  __shared__ float vv[1152];
  __shared__ float red[256];
  int tid = threadIdx.x;
  float c = rsqrtf((float)rows);
  float ss = 0.f;
  for (int j = tid; j < cols; j += 256){
    float s = 0.f;
    for (int i = 0; i < rows; ++i) s += w[i*cols + j];
    float vj = s * c;          // (M^T u0)_j
    vv[j] = vj;
    ss += vj * vj;
  }
  red[tid] = ss; __syncthreads();
  for (int o = 128; o; o >>= 1){ if (tid < o) red[tid] += red[tid + o]; __syncthreads(); }
  float inv_vn = 1.f / (sqrtf(red[0]) + 1e-12f);
  __syncthreads();
  float ts = 0.f;
  for (int i = tid; i < rows; i += 256){
    float t = 0.f;
    for (int j = 0; j < cols; ++j) t += w[i*cols + j] * vv[j];
    t *= inv_vn;               // (M v)_i
    ts += t * t;
  }
  red[tid] = ts; __syncthreads();
  for (int o = 128; o; o >>= 1){ if (tid < o) red[tid] += red[tid + o]; __syncthreads(); }
  if (tid == 0){
    float s2 = red[0];
    float sigma = s2 / (sqrtf(s2) + 1e-12f);   // u.(Mv) with u = l2(Mv)
    inv_sigma[0] = 1.f / sigma;
  }
}

// weight -> bf16 [k][cout], k = (ci*KH+ky)*KW+kx, zero-padded to Kpad
__global__ void wprep_kernel(const float* __restrict__ w, const float* __restrict__ inv_sigma,
                             u16* __restrict__ wb, int Cout, int Cin, int KH, int KW,
                             int Kdim, int Kpad, int mode){
  int t = blockIdx.x * 256 + threadIdx.x;
  int total = Kpad * Cout;
  if (t >= total) return;
  int kk = t / Cout, co = t - kk * Cout;
  float v = 0.f;
  if (kk < Kdim){
    int khw = KH * KW;
    int ci = kk / khw; int r2 = kk - ci * khw;
    int ky = r2 / KW;  int kx = r2 - ky * KW;
    int src = (mode == 0) ? (((co * Cin + ci) * KH + ky) * KW + kx)   // conv (O,I,kh,kw)
                          : (((ci * Cout + co) * KH + ky) * KW + kx); // convT (I,O,kh,kw)
    v = w[src] * inv_sigma[0];
  }
  wb[t] = f2bf(v);
}

// ---------------------------------------------- implicit-GEMM conv (WMMA)
__launch_bounds__(256)
__global__ void conv_wmma_kernel(const u16* __restrict__ in, const u16* __restrict__ wb,
    const float* __restrict__ bias, const float* __restrict__ addsrc,
    float* __restrict__ out,
    int B, int Cin, int Hin, int Win, int Cout, int Hout, int Wout,
    int KH, int KW, int stride, int pad, int Kdim, int Kpad,
    int relu_out, int mode){
  __shared__ u16 As[128 * 32];     // [pixel][k]
  __shared__ u16 Bs[64 * 32];      // [cout][k]
  __shared__ int   pxb[128];
  __shared__ short pxy[128], pxx[128];
  __shared__ short kci[32], kky[32], kkx[32];

  const int tid = threadIdx.x;
  const int lane = tid & 31;
  const int wid = tid >> 5;
  const int wave_m = wid & 3;
  const int wave_n = wid >> 2;
  const int lane15 = lane & 15;
  const int kh = lane >> 4;
  const int HWo = Hout * Wout;
  const int P = B * HWo;
  const int p0 = blockIdx.x * 128;
  const int co0 = blockIdx.y * 64;

  if (tid < 128){
    int p = p0 + tid;
    int b = 0, y = 0, x = 0;
    if (p < P){ b = p / HWo; int rem = p - b * HWo; y = rem / Wout; x = rem - y * Wout; }
    pxb[tid] = b; pxy[tid] = (short)y; pxx[tid] = (short)x;
  }

  v8f acc[2][2];
  v8f zz = {0.f,0.f,0.f,0.f,0.f,0.f,0.f,0.f};
  acc[0][0] = zz; acc[0][1] = zz; acc[1][0] = zz; acc[1][1] = zz;

  const unsigned int* A32 = (const unsigned int*)As;
  const unsigned int* B32 = (const unsigned int*)Bs;

  for (int kk0 = 0; kk0 < Kpad; kk0 += 32){
    __syncthreads();
    if (tid < 32){
      int kk = kk0 + tid;
      int ci = 0, ky = 0, kx = 0;
      if (kk < Kdim){ int khw = KH * KW; ci = kk / khw; int r2 = kk - ci * khw;
                      ky = r2 / KW; kx = r2 - ky * KW; }
      kci[tid] = (short)ci; kky[tid] = (short)ky; kkx[tid] = (short)kx;
    }
    __syncthreads();
    // im2col gather: 128x32 bf16
    #pragma unroll 4
    for (int i = 0; i < 16; ++i){
      int flat = i * 256 + tid;
      int lp = flat >> 5, kz = flat & 31;
      u16 v = 0;
      int p = p0 + lp, kk = kk0 + kz;
      if (p < P && kk < Kdim){
        int b = pxb[lp], y = pxy[lp], x = pxx[lp];
        int ci = kci[kz], ky = kky[kz], kx = kkx[kz];
        int iy, ix; bool ok;
        if (mode == 0){
          iy = y * stride - pad + ky; ix = x * stride - pad + kx;
          ok = (iy >= 0) && (iy < Hin) && (ix >= 0) && (ix < Win);
        } else {                      // transposed conv (lhs dilation)
          int ty = y + pad - ky, tx = x + pad - kx;
          iy = ty / stride; ix = tx / stride;
          ok = (ty >= 0) && (tx >= 0) && (ty - iy * stride == 0) &&
               (tx - ix * stride == 0) && (iy < Hin) && (ix < Win);
        }
        if (ok) v = in[((b * Cin + ci) * Hin + iy) * Win + ix];
      }
      As[flat] = v;
    }
    // weight tile 64x32 bf16, stored [n][k]
    #pragma unroll 2
    for (int i = 0; i < 8; ++i){
      int flat = i * 256 + tid;
      int n = flat >> 5, kz = flat & 31;
      int co = co0 + n;
      Bs[flat] = (co < Cout) ? wb[(kk0 + kz) * Cout + co] : (u16)0;
    }
    if (tid == 0 && kk0 + 32 < Kpad)
      __builtin_prefetch(wb + (size_t)(kk0 + 32) * Cout + co0, 0, 1);
    __syncthreads();

    FragBF fa[2], fb[2];
    #pragma unroll
    for (int s = 0; s < 2; ++s){
      int mrow = wave_m * 32 + s * 16 + lane15;
      int ro = mrow * 32;
      #pragma unroll
      for (int t = 0; t < 8; ++t){
        int kb = 2 * t + 16 * kh;                           // A: k = 16*kh + e
        fa[s].u[t] = A32[(ro + kb) >> 1];
      }
      int nrow = wave_n * 32 + s * 16 + lane15;
      int bo = (nrow * 32 + 16 * kh) >> 1;                  // B: k = 16*kh + e
      #pragma unroll
      for (int t = 0; t < 8; ++t) fb[s].u[t] = B32[bo + t];
    }
    #pragma unroll
    for (int mi = 0; mi < 2; ++mi)
      #pragma unroll
      for (int ni = 0; ni < 2; ++ni)
        acc[mi][ni] = __builtin_amdgcn_wmma_f32_16x16x32_bf16(
            false, fa[mi].v, false, fb[ni].v, (short)0, acc[mi][ni], false, false);
  }

  __syncthreads();
  #pragma unroll
  for (int mi = 0; mi < 2; ++mi)
    #pragma unroll
    for (int ni = 0; ni < 2; ++ni)
      #pragma unroll
      for (int r = 0; r < 8; ++r){
        int m_loc = wave_m * 32 + mi * 16 + r + 8 * kh;
        int n_loc = wave_n * 32 + ni * 16 + lane15;
        int p = p0 + m_loc, co = co0 + n_loc;
        if (p < P && co < Cout){
          float v = acc[mi][ni][r];
          if (bias) v += bias[co];
          int b = pxb[m_loc], y = pxy[m_loc], x = pxx[m_loc];
          int o = ((b * Cout + co) * Hout + y) * Wout + x;
          if (addsrc) v += addsrc[o];
          if (relu_out) v = fmaxf(v, 0.f);
          out[o] = v;
        }
      }
}

// --------------------------------------------- VQ argmin + loss + histogram
__launch_bounds__(256)
__global__ void vq_argmin_kernel(const float* __restrict__ z, const float* __restrict__ cb,
                                 int* __restrict__ idx_out, float* __restrict__ sumsq,
                                 int* __restrict__ counts){
  __shared__ float cbs[128 * 64];   // codebook chunk (32 KB)
  __shared__ float cns[128];
  __shared__ float wl[8];
  const int tid = threadIdx.x;
  const int lane = tid & 31;
  const int wid = tid >> 5;
  const int lane15 = lane & 15;
  const int kh = lane >> 4;
  const int gp = blockIdx.x * 128 + wid * 16 + lane15;   // pixel (A-matrix row)
  const int b = gp >> 8, rem = gp & 255;                 // z: [B,64,16,16]

  v2f a2[16];                                            // this lane's half of z-vec
  #pragma unroll
  for (int f = 0; f < 16; ++f){
    int k = 4 * f + 2 * kh;
    a2[f].x = z[(b * 64 + k) * 256 + rem];
    a2[f].y = z[(b * 64 + k + 1) * 256 + rem];
  }
  float best[8]; int bidx[8];
  #pragma unroll
  for (int r = 0; r < 8; ++r){ best[r] = 3.4e38f; bidx[r] = 0; }

  for (int c = 0; c < 4; ++c){       // 4 chunks x 128 codes
    __syncthreads();
    // async global->LDS staging of the 32KB codebook chunk:
    // 256 lanes x 8 iters x B128 = 2048 x 16B transfers on the ASYNCcnt path.
    {
      const float* src = cb + c * 8192;
      #pragma unroll
      for (int i = 0; i < 8; ++i){
        int felt = (i * 256 + tid) * 4;                     // float index of this b128
        unsigned ldsa = (unsigned)(uintptr_t)(&cbs[felt]);  // LDS byte addr = addr[31:0]
        unsigned goff = (unsigned)(felt * 4);               // global byte offset
        asm volatile("global_load_async_to_lds_b128 %0, %1, %2"
                     :: "v"(ldsa), "v"(goff), "s"(src) : "memory");
      }
      asm volatile("s_wait_asynccnt 0" ::: "memory");
    }
    __syncthreads();
    if (tid < 128){
      float s = 0.f;
      for (int j = 0; j < 64; ++j){ float u = cbs[tid * 64 + j]; s += u * u; }
      cns[tid] = s;
    }
    __syncthreads();
    for (int t = 0; t < 8; ++t){
      v8f acc = {0.f,0.f,0.f,0.f,0.f,0.f,0.f,0.f};
      int nrow = t * 16 + lane15;
      #pragma unroll
      for (int f = 0; f < 16; ++f){
        v2f bb;
        int k = 4 * f + 2 * kh;
        bb.x = cbs[nrow * 64 + k];
        bb.y = cbs[nrow * 64 + k + 1];
        acc = __builtin_amdgcn_wmma_f32_16x16x4_f32(
            false, a2[f], false, bb, (short)0, acc, false, false);
      }
      float cn = cns[nrow];
      int ng = c * 128 + nrow;
      #pragma unroll
      for (int r = 0; r < 8; ++r){
        float d = cn - 2.f * acc[r];          // |c|^2 - 2 z.c  (|z|^2 const)
        if (d < best[r]){ best[r] = d; bidx[r] = ng; }
      }
    }
  }
  // min over the 16 lanes that share a pixel (tie -> lowest index)
  #pragma unroll
  for (int r = 0; r < 8; ++r){
    #pragma unroll
    for (int off = 8; off >= 1; off >>= 1){
      float ob = __shfl_xor(best[r], off, 32);
      int   oi = __shfl_xor(bidx[r], off, 32);
      if (ob < best[r] || (ob == best[r] && oi < bidx[r])){ best[r] = ob; bidx[r] = oi; }
    }
  }
  // |z|^2 per pixel (pixel == lane&15 for the A fragments)
  float pn = 0.f;
  #pragma unroll
  for (int f = 0; f < 16; ++f) pn += a2[f].x * a2[f].x + a2[f].y * a2[f].y;
  float zn_full = pn + __shfl_xor(pn, 16, 32);

  float lacc = 0.f;
  const int base = blockIdx.x * 128 + wid * 16;
  #pragma unroll
  for (int r = 0; r < 8; ++r){
    float v0 = __shfl(zn_full, r, 32);
    float v1 = __shfl(zn_full, r + 8, 32);
    if (lane == 0){
      int g = base + r;
      idx_out[g] = bidx[r];
      atomicAdd(&counts[bidx[r]], 1);
      lacc += v0 + best[r];                  // |z|^2 + |c|^2 - 2 z.c = |q - z|^2
    } else if (lane == 16){
      int g = base + r + 8;
      idx_out[g] = bidx[r];
      atomicAdd(&counts[bidx[r]], 1);
      lacc += v1 + best[r];
    }
  }
  lacc += __shfl_xor(lacc, 16, 32);
  if (lane == 0) wl[wid] = lacc;
  __syncthreads();
  if (tid == 0){
    float s = 0.f;
    for (int i = 0; i < 8; ++i) s += wl[i];
    atomicAdd(sumsq, s);
  }
}

// quant (== q, straight-through forward) in bf16 NCHW for the decoder
__global__ void vq_gather_kernel(const int* __restrict__ idx, const float* __restrict__ cb,
                                 u16* __restrict__ qb, int n){
  int t = blockIdx.x * 256 + threadIdx.x;
  if (t >= n) return;
  int b = t >> 14;
  int c = (t >> 8) & 63;
  int rem = t & 255;
  int p = b * 256 + rem;
  qb[t] = f2bf(cb[idx[p] * 64 + c]);
}

__global__ void finalize_kernel(const float* __restrict__ sumsq, const int* __restrict__ counts,
                                float* __restrict__ out, int perp_off){
  __shared__ float red[512];
  int tid = threadIdx.x;
  float p = (float)counts[tid] * (1.f / 65536.f);
  red[tid] = p * logf(p + 1e-10f);
  __syncthreads();
  for (int o = 256; o; o >>= 1){ if (tid < o) red[tid] += red[tid + o]; __syncthreads(); }
  if (tid == 0){
    out[perp_off] = expf(-red[0]);
    out[0] = 1.25f * sumsq[0] / (65536.f * 64.f);   // q_loss + 0.25*e_loss
  }
}

// ---------------------------------------------------------------- driver
extern "C" void kernel_launch(void* const* d_in, const int* in_sizes, int n_in,
                              void* d_out, int out_size, void* d_ws, size_t ws_size,
                              hipStream_t stream){
  (void)in_sizes; (void)n_in; (void)out_size; (void)ws_size;
  const float* x      = (const float*)d_in[0];
  const float* ec1w   = (const float*)d_in[1];  const float* ec1b = (const float*)d_in[2];
  const float* ec2w   = (const float*)d_in[3];  const float* ec2b = (const float*)d_in[4];
  const float* ec3w   = (const float*)d_in[5];  const float* ec3b = (const float*)d_in[6];
  const float* er0w1  = (const float*)d_in[7];  const float* er0w2 = (const float*)d_in[8];
  const float* er1w1  = (const float*)d_in[9];  const float* er1w2 = (const float*)d_in[10];
  const float* prew   = (const float*)d_in[11]; const float* preb  = (const float*)d_in[12];
  const float* cbk    = (const float*)d_in[13];
  const float* dc1w   = (const float*)d_in[14]; const float* dc1b  = (const float*)d_in[15];
  const float* dr0w1  = (const float*)d_in[16]; const float* dr0w2 = (const float*)d_in[17];
  const float* dr1w1  = (const float*)d_in[18]; const float* dr1w2 = (const float*)d_in[19];
  const float* dt1w   = (const float*)d_in[20]; const float* dt1b  = (const float*)d_in[21];
  const float* dt2w   = (const float*)d_in[22]; const float* dt2b  = (const float*)d_in[23];
  float* out = (float*)d_out;

  char* ws = (char*)d_ws;
  float* sumsq  = (float*)(ws + 0);
  float* sig    = (float*)(ws + 256);     // 15 scalars
  int*   counts = (int*)  (ws + 1024);
  int*   idx    = (int*)  (ws + 8192);    // 65536 ints
  size_t off = 8192 + 65536 * 4;
  auto allocb = [&](size_t bytes) -> char* {
    off = (off + 255) & ~(size_t)255;
    char* p = ws + off; off += bytes; return p;
  };
  u16* wb_ec1 = (u16*)allocb((size_t)192  * 32  * 2);
  u16* wb_ec2 = (u16*)allocb((size_t)512  * 128 * 2);
  u16* wb_ec3 = (u16*)allocb((size_t)1152 * 128 * 2);
  u16* wb_er01 = (u16*)allocb((size_t)1152 * 64 * 2);
  u16* wb_er02 = (u16*)allocb((size_t)64   * 128 * 2);
  u16* wb_er11 = (u16*)allocb((size_t)1152 * 64 * 2);
  u16* wb_er12 = (u16*)allocb((size_t)64   * 128 * 2);
  u16* wb_pre  = (u16*)allocb((size_t)128  * 64  * 2);
  u16* wb_dc1  = (u16*)allocb((size_t)576  * 128 * 2);
  u16* wb_dr01 = (u16*)allocb((size_t)1152 * 64 * 2);
  u16* wb_dr02 = (u16*)allocb((size_t)64   * 128 * 2);
  u16* wb_dr11 = (u16*)allocb((size_t)1152 * 64 * 2);
  u16* wb_dr12 = (u16*)allocb((size_t)64   * 128 * 2);
  u16* wb_dt1  = (u16*)allocb((size_t)2048 * 64 * 2);
  u16* wb_dt2  = (u16*)allocb((size_t)1024 * 16 * 2);
  u16*   xb  = (u16*)  allocb((size_t)11534336 * 2);
  u16*   hb0 = (u16*)  allocb((size_t)16777216 * 2);
  u16*   hb1 = (u16*)  allocb((size_t)4194304  * 2);
  float* f0  = (float*)allocb((size_t)16777216 * 4);
  float* f1  = (float*)allocb((size_t)16777216 * 4);

  zero_kernel<<<1, 512, 0, stream>>>(sumsq, counts);

  auto sn = [&](const float* w, int rows, int cols, int slot){
    sn_sigma_kernel<<<1, 256, 0, stream>>>(w, rows, cols, sig + slot);
  };
  sn(ec1w, 32, 176, 0);   sn(ec2w, 128, 512, 1);  sn(ec3w, 128, 1152, 2);
  sn(er0w1, 64, 1152, 3); sn(er0w2, 128, 64, 4);  sn(er1w1, 64, 1152, 5);
  sn(er1w2, 128, 64, 6);  sn(prew, 64, 128, 7);   sn(dc1w, 128, 576, 8);
  sn(dr0w1, 64, 1152, 9); sn(dr0w2, 128, 64, 10); sn(dr1w1, 64, 1152, 11);
  sn(dr1w2, 128, 64, 12); sn(dt1w, 128, 1024, 13); sn(dt2w, 64, 176, 14);

  auto wprep = [&](const float* w, int slot, u16* wb, int Cout, int Cin, int KH, int KW,
                   int Kdim, int Kpad, int mode){
    int tot = Kpad * Cout;
    wprep_kernel<<<(tot + 255) / 256, 256, 0, stream>>>(w, sig + slot, wb, Cout, Cin,
                                                        KH, KW, Kdim, Kpad, mode);
  };
  wprep(ec1w, 0, wb_ec1, 32, 11, 4, 4, 176, 192, 0);
  wprep(ec2w, 1, wb_ec2, 128, 32, 4, 4, 512, 512, 0);
  wprep(ec3w, 2, wb_ec3, 128, 128, 3, 3, 1152, 1152, 0);
  wprep(er0w1, 3, wb_er01, 64, 128, 3, 3, 1152, 1152, 0);
  wprep(er0w2, 4, wb_er02, 128, 64, 1, 1, 64, 64, 0);
  wprep(er1w1, 5, wb_er11, 64, 128, 3, 3, 1152, 1152, 0);
  wprep(er1w2, 6, wb_er12, 128, 64, 1, 1, 64, 64, 0);
  wprep(prew, 7, wb_pre, 64, 128, 1, 1, 128, 128, 0);
  wprep(dc1w, 8, wb_dc1, 128, 64, 3, 3, 576, 576, 0);
  wprep(dr0w1, 9, wb_dr01, 64, 128, 3, 3, 1152, 1152, 0);
  wprep(dr0w2, 10, wb_dr02, 128, 64, 1, 1, 64, 64, 0);
  wprep(dr1w1, 11, wb_dr11, 64, 128, 3, 3, 1152, 1152, 0);
  wprep(dr1w2, 12, wb_dr12, 128, 64, 1, 1, 64, 64, 0);
  wprep(dt1w, 13, wb_dt1, 64, 128, 4, 4, 2048, 2048, 1);
  wprep(dt2w, 14, wb_dt2, 11, 64, 4, 4, 1024, 1024, 1);

  auto prep = [&](const float* in, u16* ob, int n, int relu){
    prep_bf16_kernel<<<(n + 255) / 256, 256, 0, stream>>>(in, ob, n, relu);
  };
  auto conv = [&](const u16* in, const u16* wb, const float* bias, const float* add,
                  float* o, int Cin, int Hin, int Win, int Cout, int Hout, int Wout,
                  int KH, int KW, int st, int pad, int Kdim, int Kpad, int relu, int mode){
    dim3 g((256 * Hout * Wout + 127) / 128, (Cout + 63) / 64);
    conv_wmma_kernel<<<g, 256, 0, stream>>>(in, wb, bias, add, o, 256, Cin, Hin, Win,
        Cout, Hout, Wout, KH, KW, st, pad, Kdim, Kpad, relu, mode);
  };
  auto resblock = [&](float* xf, const u16* w1b, const u16* w2b){
    prep(xf, hb0, 8388608, 1);
    conv(hb0, w1b, nullptr, nullptr, f1, 128, 16, 16, 64, 16, 16, 3, 3, 1, 1, 1152, 1152, 0, 0);
    prep(f1, hb1, 4194304, 1);
    conv(hb1, w2b, nullptr, xf, xf, 64, 16, 16, 128, 16, 16, 1, 1, 1, 0, 64, 64, 0, 0);
  };

  // ---- encoder ----
  prep(x, xb, 11534336, 0);
  conv(xb, wb_ec1, ec1b, nullptr, f0, 11, 64, 64, 32, 33, 33, 4, 4, 2, 2, 176, 192, 1, 0);
  prep(f0, hb0, 8921088, 0);
  conv(hb0, wb_ec2, ec2b, nullptr, f1, 32, 33, 33, 128, 16, 16, 4, 4, 2, 1, 512, 512, 1, 0);
  prep(f1, hb0, 8388608, 0);
  conv(hb0, wb_ec3, ec3b, nullptr, f0, 128, 16, 16, 128, 16, 16, 3, 3, 1, 1, 1152, 1152, 0, 0);
  resblock(f0, wb_er01, wb_er02);
  resblock(f0, wb_er11, wb_er12);
  prep(f0, hb0, 8388608, 1);                       // final ReLU of res stack
  conv(hb0, wb_pre, preb, nullptr, f1, 128, 16, 16, 64, 16, 16, 1, 1, 1, 0, 128, 128, 0, 0);

  // ---- vector quantizer ----
  vq_argmin_kernel<<<512, 256, 0, stream>>>(f1, cbk, idx, sumsq, counts);
  vq_gather_kernel<<<(4194304 + 255) / 256, 256, 0, stream>>>(idx, cbk, hb0, 4194304);

  // ---- decoder ----
  conv(hb0, wb_dc1, dc1b, nullptr, f0, 64, 16, 16, 128, 16, 16, 3, 3, 1, 1, 576, 576, 0, 0);
  resblock(f0, wb_dr01, wb_dr02);
  resblock(f0, wb_dr11, wb_dr12);
  prep(f0, hb0, 8388608, 1);
  conv(hb0, wb_dt1, dt1b, nullptr, f1, 128, 16, 16, 64, 32, 32, 4, 4, 2, 1, 2048, 2048, 1, 1);
  prep(f1, hb0, 16777216, 0);
  conv(hb0, wb_dt2, dt2b, nullptr, out + 1, 64, 32, 32, 11, 62, 62, 4, 4, 2, 2, 1024, 1024, 0, 1);

  finalize_kernel<<<1, 512, 0, stream>>>(sumsq, counts, out, 1 + 10824704);
}
